// MultiHeadAttention_71270687310307
// MI455X (gfx1250) — compile-verified, matrix operand order checked by reading, not run on hardware
//
#include <hip/hip_runtime.h>
#include <hip/hip_bf16.h>

typedef __bf16 bf16;
typedef bf16  v16bf __attribute__((ext_vector_type(16)));
typedef float v8f   __attribute__((ext_vector_type(8)));

#define DIM   1024
#define NHEAD 16
#define HDIM  64
#define BATCH 4
#define SEQ   2048
#define MTOT  (BATCH * SEQ)   // 8192 rows

// ---------------------------------------------------------------------------
// Fragment load: 16-bit A/B operand, 16x32 tile striped per CDNA5 WMMA layout.
// Lane L holds row (L&15); K-halves: lanes 0-15 get k = 0..7 & 16..23,
// lanes 16-31 get k = 8..15 & 24..31  ->  two contiguous 16-byte loads.
// ---------------------------------------------------------------------------
__device__ __forceinline__ v16bf frag_ld(const bf16* p) {
    union { uint4 q[2]; v16bf v; } u;
    u.q[0] = *(const uint4*)p;
    u.q[1] = *(const uint4*)(p + 16);
    return u.v;
}

__device__ __forceinline__ v8f wmma_bf16(v16bf a, v16bf b, v8f c) {
    return __builtin_amdgcn_wmma_f32_16x16x32_bf16(
        /*neg_a=*/false, a, /*neg_b=*/false, b,
        /*c_mod=*/(short)0, c, /*reuse_a=*/false, /*reuse_b=*/false);
}

// ---------------------------------------------------------------------------
// CDNA5 async global->LDS copy (GV mode): per-lane 16-byte transfer, tracked
// by ASYNCcnt; LDS byte address = low 32 bits of generic shared pointer.
// ---------------------------------------------------------------------------
__device__ __forceinline__ void async_copy16(void* lds_dst, const void* gsrc) {
    unsigned lds = (unsigned)(size_t)lds_dst;
    asm volatile("global_load_async_to_lds_b128 %0, %1, off"
                 :: "v"(lds), "v"(gsrc)
                 : "memory");
}
__device__ __forceinline__ void wait_async() {
    asm volatile("s_wait_asynccnt 0" ::: "memory");
}

// Raw v_exp_f32 / v_rcp_f32 (no libm denorm/range fixup sequences).
__device__ __forceinline__ float fast_exp2(float x) { return __builtin_amdgcn_exp2f(x); }
__device__ __forceinline__ float fast_rcp(float x)  { return __builtin_amdgcn_rcpf(x); }

// ---------------------------------------------------------------------------
// f32 -> bf16 conversion (n divisible by 4)
// ---------------------------------------------------------------------------
__global__ void f32_to_bf16(const float* __restrict__ src, bf16* __restrict__ dst, int n) {
    int i = (blockIdx.x * blockDim.x + threadIdx.x) * 4;
    if (i + 4 <= n) {
        float4 f = *(const float4*)(src + i);
        dst[i]     = (bf16)f.x;
        dst[i + 1] = (bf16)f.y;
        dst[i + 2] = (bf16)f.z;
        dst[i + 3] = (bf16)f.w;
    }
}

// ---------------------------------------------------------------------------
// NT GEMM: C[M,N] = A[M,K] * B[N,K]^T.
// MODE 0: bf16 out [M,N];  MODE 1: bf16 out transposed to [b, n, s] (SEQ rows);
// MODE 2: f32 out [M,N] + bias.
// 256 threads = 8 wave32; block tile 128x64; each wave 32x32 (2x2 WMMA tiles).
// ---------------------------------------------------------------------------
template <int MODE>
__global__ __launch_bounds__(256)
void gemm_nt(const bf16* __restrict__ A, const bf16* __restrict__ B,
             bf16* __restrict__ Cb, float* __restrict__ Cf,
             const float* __restrict__ bias, int M, int N, int K)
{
    __shared__ alignas(16) bf16 As[128 * 32];
    __shared__ alignas(16) bf16 Bs[64 * 32];

    const int tid  = threadIdx.x;
    const int lane = tid & 31;
    const int wave = tid >> 5;
    const int wm   = wave >> 1;          // 0..3  (M direction)
    const int wn   = wave & 1;           // 0..1  (N direction)
    const int m0   = blockIdx.y * 128;
    const int n0   = blockIdx.x * 64;
    const int row  = lane & 15;
    const int kb   = (lane < 16) ? 0 : 8;

    v8f acc[2][2] = {};

    for (int k0 = 0; k0 < K; k0 += 32) {
        __syncthreads();
        {   // stage A tile 128x32: each thread 32B via two async b128
            int r = tid >> 1, c = (tid & 1) * 16;
            const bf16* src = &A[(size_t)(m0 + r) * K + k0 + c];
            async_copy16(&As[r * 32 + c],     src);
            async_copy16(&As[r * 32 + c + 8], src + 8);
        }
        {   // stage B tile 64x32: each thread 16B
            int r = tid >> 2, c = (tid & 3) * 8;
            async_copy16(&Bs[r * 32 + c], &B[(size_t)(n0 + r) * K + k0 + c]);
        }
        // hint next A tile into cache (speculative; dropped if OOB)
        __builtin_prefetch(&A[(size_t)(m0 + (tid >> 2)) * K + k0 + 32], 0, 0);
        wait_async();
        __syncthreads();

        v16bf af[2], bfr[2];
#pragma unroll
        for (int i = 0; i < 2; ++i)
            af[i] = frag_ld(&As[(wm * 32 + i * 16 + row) * 32 + kb]);
#pragma unroll
        for (int j = 0; j < 2; ++j)
            bfr[j] = frag_ld(&Bs[(wn * 32 + j * 16 + row) * 32 + kb]);
#pragma unroll
        for (int i = 0; i < 2; ++i)
#pragma unroll
            for (int j = 0; j < 2; ++j)
                acc[i][j] = wmma_bf16(af[i], bfr[j], acc[i][j]);
    }

    // C/D layout: VGPR r, lanes 0-15 -> M=r, lanes 16-31 -> M=r+8; N = lane&15
    const int hi = (lane >> 4) * 8;
#pragma unroll
    for (int i = 0; i < 2; ++i) {
#pragma unroll
        for (int j = 0; j < 2; ++j) {
            int n = n0 + wn * 32 + j * 16 + (lane & 15);
            float bv = (MODE == 2) ? bias[n] : 0.0f;
#pragma unroll
            for (int r = 0; r < 8; ++r) {
                int m = m0 + wm * 32 + i * 16 + r + hi;
                float v = acc[i][j][r] + bv;
                if (MODE == 0) {
                    Cb[(size_t)m * N + n] = (bf16)v;
                } else if (MODE == 1) {
                    int bb = m >> 11, s = m & (SEQ - 1);
                    Cb[((size_t)bb * N + n) * SEQ + s] = (bf16)v;
                } else {
                    Cf[(size_t)m * N + n] = v;
                }
            }
        }
    }
}

// ---------------------------------------------------------------------------
// Flash attention: grid (SEQ/64, BATCH*NHEAD), 128 threads (4 waves).
// Each wave owns 16 query rows. Scores computed TRANSPOSED (S^T = K * Q^T) so
// softmax row stats reduce with a single shfl_xor(16) per block.
// Q,K layouts: [b, s, h*64+e]; V pre-transposed to [b, h*64+e, s]. Output
// O: [b, s, h*64+e], all bf16.
// ---------------------------------------------------------------------------
__global__ __launch_bounds__(128)
void flash_attn(const bf16* __restrict__ Q, const bf16* __restrict__ Kg,
                const bf16* __restrict__ Vt_g, bf16* __restrict__ O)
{
    __shared__ alignas(16) bf16 Ks[32 * 64];       // [t, e]
    __shared__ alignas(16) bf16 Vs[64 * 32];       // [e, t]
    __shared__ alignas(16) bf16 Ps[4][16 * 32];    // per-wave prob tile [s, t]

    const int tid  = threadIdx.x;
    const int lane = tid & 31;
    const int wave = tid >> 5;                     // 0..3
    const int bh   = blockIdx.y;                   // b*NHEAD + h
    const int b    = bh >> 4;
    const int h    = bh & 15;
    const int q0   = blockIdx.x * 64 + wave * 16;
    const int row  = lane & 15;
    const int kb   = (lane < 16) ? 0 : 8;
    const int hi   = (lane >> 4) * 8;

    const size_t baseQ = (size_t)b * SEQ * DIM + (size_t)h * HDIM;
    const size_t baseV = ((size_t)b * DIM + (size_t)h * HDIM) * SEQ;

    // Q fragments (B operand of S^T = K*Q^T), kept in registers
    v16bf qf[2];
#pragma unroll
    for (int s = 0; s < 2; ++s)
        qf[s] = frag_ld(&Q[baseQ + (size_t)(q0 + row) * DIM + s * 32 + kb]);

    const float c2 = 0.125f * 1.44269504088896340736f; // 1/sqrt(HDIM) * log2(e)

    v8f o[4] = {};
    float m_s = -__builtin_inff();   // running max for query s = lane&15
    float l_s = 0.0f;                // running sum for query s = lane&15

    for (int t0 = 0; t0 < SEQ; t0 += 32) {
        __syncthreads();   // previous iteration's LDS reads done
        {   // stage K tile [32 x 64]: each thread 32B via two async b128
            int r = tid >> 2, c = (tid & 3) * 16;
            const bf16* src = &Kg[baseQ + (size_t)(t0 + r) * DIM + c];
            async_copy16(&Ks[r * 64 + c],     src);
            async_copy16(&Ks[r * 64 + c + 8], src + 8);
        }
        {   // stage V^T tile [64 x 32] from pre-transposed V
            int r = tid >> 1, c = (tid & 1) * 16;
            const bf16* src = &Vt_g[baseV + (size_t)r * SEQ + t0 + c];
            async_copy16(&Vs[r * 32 + c],     src);
            async_copy16(&Vs[r * 32 + c + 8], src + 8);
        }
        wait_async();
        __syncthreads();

        // S^T tiles: st0 -> t in [0,16), st1 -> t in [16,32); lane elem r
        // holds (t = tile*16 + hi + r, s = lane&15)
        v8f st0 = {}, st1 = {};
#pragma unroll
        for (int s = 0; s < 2; ++s) {
            v16bf k0f = frag_ld(&Ks[(row) * 64 + s * 32 + kb]);
            v16bf k1f = frag_ld(&Ks[(16 + row) * 64 + s * 32 + kb]);
            st0 = wmma_bf16(k0f, qf[s], st0);
            st1 = wmma_bf16(k1f, qf[s], st1);
        }

        // online softmax for query column s = lane&15 (exp2 domain)
        float a0[8], a1[8];
        float mloc = -__builtin_inff();
#pragma unroll
        for (int r = 0; r < 8; ++r) {
            a0[r] = st0[r] * c2;
            a1[r] = st1[r] * c2;
            mloc = fmaxf(mloc, fmaxf(a0[r], a1[r]));
        }
        mloc = fmaxf(mloc, __shfl_xor(mloc, 16, 32));   // combine t-halves
        float mnew  = fmaxf(m_s, mloc);
        float alpha = fast_exp2(m_s - mnew);

        union { bf16 h[8]; uint4 q; } p0, p1;
        float sum = 0.0f;
#pragma unroll
        for (int r = 0; r < 8; ++r) {
            float x0 = fast_exp2(a0[r] - mnew);
            float x1 = fast_exp2(a1[r] - mnew);
            p0.h[r] = (bf16)x0;
            p1.h[r] = (bf16)x1;
            sum += x0 + x1;
        }
        sum += __shfl_xor(sum, 16, 32);
        l_s = l_s * alpha + sum;
        m_s = mnew;

        // rescale O (O rows s = r + hi; fetch alpha from stat lane r+hi)
#pragma unroll
        for (int r = 0; r < 8; ++r) {
            float ar = __shfl(alpha, r + hi, 32);
#pragma unroll
            for (int j = 0; j < 4; ++j) o[j][r] *= ar;
        }

        // store P tile [s, t]: lane covers t = hi..hi+7 and 16+hi..16+hi+7
        *(uint4*)&Ps[wave][(lane & 15) * 32 + hi]      = p0.q;
        *(uint4*)&Ps[wave][(lane & 15) * 32 + 16 + hi] = p1.q;
        __syncthreads();

        // O += P (16s x 32t) x V^T-tiles (B operand rows = e)
        v16bf pf = frag_ld(&Ps[wave][row * 32 + kb]);
#pragma unroll
        for (int j = 0; j < 4; ++j) {
            v16bf vf = frag_ld(&Vs[(j * 16 + row) * 32 + kb]);
            o[j] = wmma_bf16(pf, vf, o[j]);
        }
    }

    // normalize and write attention output (bf16, [b, s, h*64+e])
    float inv = fast_rcp(l_s);
#pragma unroll
    for (int r = 0; r < 8; ++r) {
        float ir = __shfl(inv, r + hi, 32);
        size_t orow = baseQ + (size_t)(q0 + r + hi) * DIM;
#pragma unroll
        for (int j = 0; j < 4; ++j)
            O[orow + j * 16 + (lane & 15)] = (bf16)(o[j][r] * ir);
    }
}

// ---------------------------------------------------------------------------
// launch
// ---------------------------------------------------------------------------
extern "C" void kernel_launch(void* const* d_in, const int* in_sizes, int n_in,
                              void* d_out, int out_size, void* d_ws, size_t ws_size,
                              hipStream_t stream)
{
    const float* x  = (const float*)d_in[0];
    const float* Wq = (const float*)d_in[1];
    const float* Wk = (const float*)d_in[2];
    const float* Wv = (const float*)d_in[3];
    const float* Wo = (const float*)d_in[4];
    const float* bo = (const float*)d_in[5];
    float* out = (float*)d_out;

    const size_t XN = (size_t)MTOT * DIM;   // 8,388,608 elems
    const size_t WN = (size_t)DIM * DIM;    // 1,048,576 elems

    bf16* ws  = (bf16*)d_ws;
    bf16* xb  = ws;  ws += XN;
    bf16* wqb = ws;  ws += WN;
    bf16* wkb = ws;  ws += WN;
    bf16* wvb = ws;  ws += WN;
    bf16* wob = ws;  ws += WN;
    bf16* Qb  = ws;  ws += XN;
    bf16* Kb  = ws;  ws += XN;
    bf16* VtB = ws;  ws += XN;   // V pre-transposed: [b, d, s]
    bf16* AOb = ws;  ws += XN;

    // 1) f32 -> bf16 conversions
    f32_to_bf16<<<(int)(XN / 4 + 255) / 256, 256, 0, stream>>>(x,  xb,  (int)XN);
    f32_to_bf16<<<(int)(WN / 4 + 255) / 256, 256, 0, stream>>>(Wq, wqb, (int)WN);
    f32_to_bf16<<<(int)(WN / 4 + 255) / 256, 256, 0, stream>>>(Wk, wkb, (int)WN);
    f32_to_bf16<<<(int)(WN / 4 + 255) / 256, 256, 0, stream>>>(Wv, wvb, (int)WN);
    f32_to_bf16<<<(int)(WN / 4 + 255) / 256, 256, 0, stream>>>(Wo, wob, (int)WN);

    // 2) Q/K/V projections: [8192,1024] x [1024,1024]^T (heads concatenated in N);
    //    V written transposed to [b, d, s] for the flash inner loop.
    dim3 gg(DIM / 64, MTOT / 128);  // 16 x 64
    gemm_nt<0><<<gg, 256, 0, stream>>>(xb, wqb, Qb,  nullptr, nullptr, MTOT, DIM, DIM);
    gemm_nt<0><<<gg, 256, 0, stream>>>(xb, wkb, Kb,  nullptr, nullptr, MTOT, DIM, DIM);
    gemm_nt<1><<<gg, 256, 0, stream>>>(xb, wvb, VtB, nullptr, nullptr, MTOT, DIM, DIM);

    // 3) flash attention per (b,h)
    dim3 ga(SEQ / 64, BATCH * NHEAD);  // 32 x 64
    flash_attn<<<ga, 128, 0, stream>>>(Qb, Kb, VtB, AOb);

    // 4) output projection + bias -> f32 d_out
    gemm_nt<2><<<gg, 256, 0, stream>>>(AOb, wob, nullptr, out, bo, MTOT, DIM, DIM);
}
